// st_gcn_short_6476810682377
// MI455X (gfx1250) — matrix operand order repaired and checked by reading, not compiled, not run on hardware
//
#include <hip/hip_runtime.h>
#include <hip/hip_bf16.h>

typedef __attribute__((ext_vector_type(16))) __bf16 v16bf;
typedef __attribute__((ext_vector_type(8)))  __bf16 v8bf;
typedef __attribute__((ext_vector_type(8)))  float  v8f;

// CDNA5 async global->LDS copy support (guarded; falls back to sync staging)
#if defined(__has_builtin)
#if __has_builtin(__builtin_amdgcn_global_load_async_to_lds_b128) && \
    __has_builtin(__builtin_amdgcn_s_wait_asynccnt)
#define CDNA5_HAS_ASYNC_LDS 1
#endif
#endif
#ifndef CDNA5_HAS_ASYNC_LDS
#define CDNA5_HAS_ASYNC_LDS 0
#endif

#if CDNA5_HAS_ASYNC_LDS
#define GLOBAL_AS __attribute__((address_space(1)))
#define LDS_AS    __attribute__((address_space(3)))
typedef int v4i_ __attribute__((vector_size(16)));
__device__ __forceinline__ void async_copy_b128(const void* g, void* l) {
    __builtin_amdgcn_global_load_async_to_lds_b128(
        (GLOBAL_AS v4i_*)g, (LDS_AS v4i_*)l, 0, 0);
}
#endif

// ---------------------------------------------------------------------------
// Kernel 1: tiled transpose + f32->bf16 convert.
//   dst[col][row] = (bf16) src[row][col],  src is [R][C] row-major (per batch z)
// Used for: x [B][C=1024][TN=768] -> featB [B*768][1024]   (batch = B)
//           w  [K=1024][N=1024]   -> wT    [N=1024][K=1024] (batch = 1)
// ---------------------------------------------------------------------------
__global__ __launch_bounds__(256)
void transpose_cvt_kernel(const float* __restrict__ src, __bf16* __restrict__ dst,
                          int R, int C, long long sStride, long long dStride)
{
    __shared__ float tile[32][33];
    const int c0 = blockIdx.x * 32;
    const int r0 = blockIdx.y * 32;
    const float* s = src + (size_t)blockIdx.z * sStride;
    __bf16*     d = dst + (size_t)blockIdx.z * dStride;
    const int tx = threadIdx.x, ty = threadIdx.y;   // 32 x 8
#pragma unroll
    for (int i = 0; i < 32; i += 8)
        tile[ty + i][tx] = s[(size_t)(r0 + ty + i) * C + c0 + tx];
    __syncthreads();
#pragma unroll
    for (int i = 0; i < 32; i += 8)
        d[(size_t)(c0 + ty + i) * R + r0 + tx] = (__bf16)tile[tx][ty + i];
}

// ---------------------------------------------------------------------------
// Kernel 2: fused triple GEMM  [12288 x 1024] x [1024 x 1024] for
//   mat 0: theta = F@Wth + b_theta  -> bf16
//   mat 1: phi   = F@Wph + b_phi    -> bf16
//   mat 2: FW    = F@Wgc            -> f32
// Block tile 128x128, BK=32, 256 threads = 8 waves, each wave 64x32
// (4 M-tiles x 2 N-tiles of 16x16), v_wmma_f32_16x16x32_bf16.
// Double-buffered LDS with GLOBAL_LOAD_ASYNC_TO_LDS_B128 pipelining.
// ---------------------------------------------------------------------------
#define BM 128
#define BN 128
#define BK 32
#define KSTEPS 32   // 1024 / BK

__global__ __launch_bounds__(256)
void gemm3_kernel(const __bf16* __restrict__ featB,
                  const __bf16* __restrict__ wthT, const __bf16* __restrict__ wphT,
                  const __bf16* __restrict__ wgcT,
                  const float* __restrict__ b_theta, const float* __restrict__ b_phi,
                  __bf16* __restrict__ thetaB, __bf16* __restrict__ phiB,
                  float* __restrict__ fwF)
{
    __shared__ __align__(16) __bf16 As[2][BM][BK];   // As[buf][m][k]
    __shared__ __align__(16) __bf16 Bs[2][BN][BK];   // Bs[buf][n][k] (col-major tile)

    const int tid  = threadIdx.x;
    const int lane = tid & 31;
    const int wave = tid >> 5;
    const int wm   = wave & 1;      // 0..1  (64-row stripes)
    const int wn   = wave >> 1;     // 0..3  (32-col stripes)
    const int h    = lane >> 4;     // half-wave select
    const int l15  = lane & 15;

    const int mat   = blockIdx.y >> 3;            // 0=theta 1=phi 2=gcn
    const int ncol0 = (blockIdx.y & 7) * BN;      // col base inside 1024
    const int row0  = blockIdx.x * BM;

    const __bf16* wT = (mat == 0) ? wthT : (mat == 1) ? wphT : wgcT;

    v8f acc[4][2];
#pragma unroll
    for (int i = 0; i < 4; ++i)
#pragma unroll
        for (int j = 0; j < 2; ++j)
#pragma unroll
            for (int e = 0; e < 8; ++e) acc[i][j][e] = 0.0f;

    const int lm = tid >> 1;     // 0..127 staged row
    const int lh = tid & 1;      // which 16-element half of BK

    const __bf16* gA = featB + (size_t)(row0 + lm) * 1024 + lh * 16;
    const __bf16* gB = wT    + (size_t)(ncol0 + lm) * 1024 + lh * 16;

    auto stage = [&](int buf, int k0) {
        const __bf16* ga = gA + k0;
        const __bf16* gb = gB + k0;
        __bf16* la = &As[buf][lm][lh * 16];
        __bf16* lb = &Bs[buf][lm][lh * 16];
#if CDNA5_HAS_ASYNC_LDS
        async_copy_b128(ga,     la);
        async_copy_b128(ga + 8, la + 8);
        async_copy_b128(gb,     lb);
        async_copy_b128(gb + 8, lb + 8);
#else
        const uint4* s1 = (const uint4*)ga; uint4 a0 = s1[0], a1 = s1[1];
        const uint4* s2 = (const uint4*)gb; uint4 b0 = s2[0], b1 = s2[1];
        uint4* d1 = (uint4*)la; d1[0] = a0; d1[1] = a1;
        uint4* d2 = (uint4*)lb; d2[0] = b0; d2[1] = b1;
#endif
    };

    stage(0, 0);   // prologue: fill buffer 0

    for (int kk = 0; kk < KSTEPS; ++kk) {
        const int buf = kk & 1;
#if CDNA5_HAS_ASYNC_LDS
        __builtin_amdgcn_s_wait_asynccnt(0);   // own async writes to buf done
#endif
        __syncthreads();                       // all waves: buf ready, buf^1 free
        if (kk + 1 < KSTEPS) stage(buf ^ 1, (kk + 1) * BK);

        v16bf afrag[4], bfrag[2];
#pragma unroll
        for (int mi = 0; mi < 4; ++mi) {
            // A 16x32 bf16 layout: lane-half h -> K runs [8h..8h+7], [16+8h..16+8h+7]
            const __bf16* p = &As[buf][wm * 64 + mi * 16 + l15][0];
            v8bf lo = *(const v8bf*)(p + 8 * h);
            v8bf hi = *(const v8bf*)(p + 16 + 8 * h);
            v16bf a;
#pragma unroll
            for (int e = 0; e < 8; ++e) { a[e] = lo[e]; a[e + 8] = hi[e]; }
            afrag[mi] = a;
        }
#pragma unroll
        for (int ni = 0; ni < 2; ++ni) {
            // B 32x16 bf16 layout: lane-half h -> K runs [8h..8h+7], [16+8h..16+8h+7]
            const __bf16* p = &Bs[buf][wn * 32 + ni * 16 + l15][0];
            v8bf lo = *(const v8bf*)(p + 8 * h);
            v8bf hi = *(const v8bf*)(p + 16 + 8 * h);
            v16bf b;
#pragma unroll
            for (int e = 0; e < 8; ++e) { b[e] = lo[e]; b[e + 8] = hi[e]; }
            bfrag[ni] = b;
        }
#pragma unroll
        for (int mi = 0; mi < 4; ++mi)
#pragma unroll
            for (int ni = 0; ni < 2; ++ni)
                acc[mi][ni] = __builtin_amdgcn_wmma_f32_16x16x32_bf16(
                    false, afrag[mi], false, bfrag[ni], (short)0, acc[mi][ni], false, false);
    }

    // Epilogue: D layout = VGPR r -> row r + 8*h, col = l15. Branch on mat ONCE.
    if (mat == 2) {
#pragma unroll
        for (int mi = 0; mi < 4; ++mi)
#pragma unroll
            for (int ni = 0; ni < 2; ++ni) {
                const int nG = ncol0 + wn * 32 + ni * 16 + l15;
#pragma unroll
                for (int r = 0; r < 8; ++r) {
                    const int mG = row0 + wm * 64 + mi * 16 + 8 * h + r;
                    fwF[(size_t)mG * 1024 + nG] = acc[mi][ni][r];
                }
            }
    } else {
        __bf16* dst        = (mat == 0) ? thetaB : phiB;
        const float* bias  = (mat == 0) ? b_theta : b_phi;
#pragma unroll
        for (int ni = 0; ni < 2; ++ni) {
            const int nG = ncol0 + wn * 32 + ni * 16 + l15;
            const float bv = bias[nG];
#pragma unroll
            for (int mi = 0; mi < 4; ++mi)
#pragma unroll
                for (int r = 0; r < 8; ++r) {
                    const int mG = row0 + wm * 64 + mi * 16 + 8 * h + r;
                    dst[(size_t)mG * 1024 + nG] = (__bf16)(acc[mi][ni][r] + bv);
                }
        }
    }
}

// ---------------------------------------------------------------------------
// Kernel 3: per-frame fusion. One block per bt (1024 blocks, 256 threads):
//   logits = theta@phi^T/32 with position mask -> softmax -> R (also output),
//   agg = R @ FW, frame-wide LayerNorm over 12x1024, ReLU -> output.
// ---------------------------------------------------------------------------
__global__ __launch_bounds__(256)
void fuse_kernel(const __bf16* __restrict__ thetaB, const __bf16* __restrict__ phiB,
                 const float* __restrict__ fwF, const float* __restrict__ boxes,
                 const float* __restrict__ ln_w, const float* __restrict__ ln_b,
                 float* __restrict__ outMain, float* __restrict__ outR)
{
    __shared__ float px[12], py[12];
    __shared__ float lg[12][12];
    __shared__ float Rg[12][12];
    __shared__ float aggS[12 * 1024];
    __shared__ float s_sum, s_sq;

    const int bt  = blockIdx.x;
    const int tid = threadIdx.x;

    if (tid < 12) {
        const float* bx = boxes + (size_t)bt * 48 + tid * 4;
        px[tid] = (bx[0] + bx[2]) * 0.5f;
        py[tid] = (bx[1] + bx[3]) * 0.5f;
    }
    if (tid == 12) { s_sum = 0.0f; s_sq = 0.0f; }
    __syncthreads();

    if (tid < 144) {
        const int n = tid / 12, m = tid % 12;
        const float sqn = px[n] * px[n] + py[n] * py[n];
        const float sqm = px[m] * px[m] + py[m] * py[m];
        const float dot = px[n] * px[m] + py[n] * py[m];
        const float d2  = sqn - 2.0f * dot + sqm;
        const float dist = sqrtf(fmaxf(d2, 0.0f));
        float logit;
        if (dist > 0.4f) {          // POS_THRESHOLD * OW
            logit = -3.0e38f;       // exp underflows to exactly 0
        } else {
            const __bf16* tr = thetaB + ((size_t)bt * 12 + n) * 1024;
            const __bf16* pr = phiB  + ((size_t)bt * 12 + m) * 1024;
            float a = 0.0f;
            for (int k = 0; k < 1024; k += 8) {
                v8bf tv = *(const v8bf*)(tr + k);
                v8bf pv = *(const v8bf*)(pr + k);
#pragma unroll
                for (int e = 0; e < 8; ++e) a += (float)tv[e] * (float)pv[e];
            }
            logit = a * 0.03125f;   // 1/sqrt(1024)
        }
        lg[n][m] = logit;
    }
    __syncthreads();

    if (tid < 12) {                 // softmax row
        float mx = -3.4e38f;
#pragma unroll
        for (int m = 0; m < 12; ++m) mx = fmaxf(mx, lg[tid][m]);
        float ex[12]; float s = 0.0f;
#pragma unroll
        for (int m = 0; m < 12; ++m) { ex[m] = expf(lg[tid][m] - mx); s += ex[m]; }
        const float inv = 1.0f / s;
#pragma unroll
        for (int m = 0; m < 12; ++m) {
            const float r = ex[m] * inv;
            Rg[tid][m] = r;
            outR[(size_t)bt * 144 + tid * 12 + m] = r;
        }
    }
    __syncthreads();

    float lsum = 0.0f, lsq = 0.0f;
    for (int c = tid; c < 1024; c += 256) {
        float fv[12];
#pragma unroll
        for (int m = 0; m < 12; ++m) fv[m] = fwF[((size_t)bt * 12 + m) * 1024 + c];
#pragma unroll
        for (int n = 0; n < 12; ++n) {
            float a = 0.0f;
#pragma unroll
            for (int m = 0; m < 12; ++m) a += Rg[n][m] * fv[m];
            aggS[n * 1024 + c] = a;
            lsum += a; lsq += a * a;
        }
    }
    atomicAdd(&s_sum, lsum);
    atomicAdd(&s_sq, lsq);
    __syncthreads();

    const float mean = s_sum * (1.0f / 12288.0f);
    const float var  = s_sq * (1.0f / 12288.0f) - mean * mean;
    const float rstd = rsqrtf(var + 1e-5f);
    for (int i = tid; i < 12288; i += 256) {
        const float v = (aggS[i] - mean) * rstd * ln_w[i] + ln_b[i];
        outMain[(size_t)bt * 12288 + i] = fmaxf(v, 0.0f);
    }
}

// ---------------------------------------------------------------------------
extern "C" void kernel_launch(void* const* d_in, const int* in_sizes, int n_in,
                              void* d_out, int out_size, void* d_ws, size_t ws_size,
                              hipStream_t stream)
{
    const float* x       = (const float*)d_in[0];  // [16,1024,64,12]
    const float* Abox    = (const float*)d_in[1];  // [16,64,12,4]
    const float* w_theta = (const float*)d_in[2];  // [1024,1024]
    const float* b_theta = (const float*)d_in[3];
    const float* w_phi   = (const float*)d_in[4];
    const float* b_phi   = (const float*)d_in[5];
    const float* w_gcn   = (const float*)d_in[6];
    const float* ln_w    = (const float*)d_in[7];  // [12,1024]
    const float* ln_b    = (const float*)d_in[8];

    char* ws = (char*)d_ws;
    __bf16* featB  = (__bf16*)(ws + 0);           // 12288x1024 bf16  (25165824 B)
    __bf16* wthT   = (__bf16*)(ws + 25165824);    // 1024x1024 bf16, N-major
    __bf16* wphT   = (__bf16*)(ws + 27262976);
    __bf16* wgcT   = (__bf16*)(ws + 29360128);
    __bf16* thetaB = (__bf16*)(ws + 31457280);    // 12288x1024 bf16
    __bf16* phiB   = (__bf16*)(ws + 56623104);    // 12288x1024 bf16
    float*  fwF    = (float*)(ws + 81788928);     // 12288x1024 f32  -> end 132120576

    dim3 tb(32, 8);
    // x: per-batch [1024 c][768 tn] -> featB [768 tn][1024 c]
    transpose_cvt_kernel<<<dim3(24, 32, 16), tb, 0, stream>>>(
        x, featB, 1024, 768, 1024LL * 768, 768LL * 1024);
    // weights [K][N] -> [N][K]
    transpose_cvt_kernel<<<dim3(32, 32, 1), tb, 0, stream>>>(w_theta, wthT, 1024, 1024, 0, 0);
    transpose_cvt_kernel<<<dim3(32, 32, 1), tb, 0, stream>>>(w_phi,   wphT, 1024, 1024, 0, 0);
    transpose_cvt_kernel<<<dim3(32, 32, 1), tb, 0, stream>>>(w_gcn,   wgcT, 1024, 1024, 0, 0);

    // 12288 x (3x1024) GEMM: grid (96 M-tiles, 24 N-tiles)
    gemm3_kernel<<<dim3(96, 24), 256, 0, stream>>>(
        featB, wthT, wphT, wgcT, b_theta, b_phi, thetaB, phiB, fwF);

    float* outMain = (float*)d_out;                    // [1024,12,1024]
    float* outR    = outMain + (size_t)12288 * 1024;   // [1024,12,12]
    fuse_kernel<<<1024, 256, 0, stream>>>(
        thetaB, phiB, fwF, Abox, ln_w, ln_b, outMain, outR);
}
